// RFATransformerBlock_77103252898104
// MI455X (gfx1250) — matrix operand order repaired and checked by reading, not compiled, not run on hardware
//
#include <hip/hip_runtime.h>
#include <cstdint>
#include <cstddef>

// ---------------------------------------------------------------------------
// Problem constants (match reference)
// ---------------------------------------------------------------------------
namespace cfg {
constexpr int B   = 2;
constexpr int S   = 2048;
constexpr int DM  = 512;
constexpr int H   = 8;
constexpr int DQK = 64;
constexpr int DV  = 64;
constexpr int NF  = 32;
constexpr int F   = 2 * NF;   // 64
constexpr int HID = 2048;
constexpr int M   = B * S;    // 4096 rows of activations
constexpr float EPS = 1e-6f;
}

// ---------------------------------------------------------------------------
// WMMA types (CDNA5 gfx1250, wave32)
// ---------------------------------------------------------------------------
typedef __bf16 v16bf __attribute__((ext_vector_type(16)));
typedef float  v8f   __attribute__((ext_vector_type(8)));

union FragBF {          // 16 bf16 = 8 VGPRs
    v16bf v;
    uint4 u[2];
};

#define LDA 72   // padded row length (bf16 elems): 144B -> 36-bank row stride

// ---------------------------------------------------------------------------
// 64x64x64 bf16 WMMA helper on LDS tiles with LDA-padded rows.
//   D(tiles of wave) += A[64x64] * B[64x64]   where Bl is "Bt" layout:
//   Bl[n*LDA + k] == B[k][n].
// Each wave owns 2 adjacent 16x16 output tiles: rows tr*16.., cols tc0*16..
// ---------------------------------------------------------------------------
__device__ __forceinline__
void wmma64(const __bf16* Al, const __bf16* Bl,
            int tr, int tc0, int half, int l16, v8f (&acc)[2]) {
#pragma unroll
    for (int kk = 0; kk < 64; kk += 32) {
        FragBF a;
        const __bf16* Ab = Al + (tr * 16 + l16) * LDA + kk;
        a.u[0] = *(const uint4*)(Ab + half * 8);
        a.u[1] = *(const uint4*)(Ab + 16 + half * 8);
#pragma unroll
        for (int t = 0; t < 2; ++t) {
            const __bf16* Bb = Bl + ((tc0 + t) * 16 + l16) * LDA + kk + half * 8;
            FragBF b;
            b.u[0] = *(const uint4*)(Bb);
            b.u[1] = *(const uint4*)(Bb + 16);
            acc[t] = __builtin_amdgcn_wmma_f32_16x16x32_bf16(
                false, a.v, false, b.v, (short)0, acc[t], false, false);
        }
    }
}

// ---------------------------------------------------------------------------
// Weight cast + transpose:  W[K,N] f32  ->  Wt[N,K] bf16
// ---------------------------------------------------------------------------
__global__ void cast_transpose_kernel(const float* __restrict__ W,
                                      __bf16* __restrict__ Wt,
                                      int K, int N) {
    int idx = blockIdx.x * blockDim.x + threadIdx.x;
    if (idx >= K * N) return;
    int n = idx / K;
    int k = idx - n * K;
    Wt[idx] = (__bf16)W[(size_t)k * N + n];
}

// ---------------------------------------------------------------------------
// RMSNorm over last dim (DM=512), f32 in -> bf16 out
// ---------------------------------------------------------------------------
__global__ __launch_bounds__(256)
void rmsnorm_kernel(const float* __restrict__ x,
                    const float* __restrict__ g,
                    __bf16* __restrict__ out) {
    __shared__ float sdata[256];
    const int row = blockIdx.x;
    const int tid = threadIdx.x;
    const float2 v = *(const float2*)(x + (size_t)row * cfg::DM + tid * 2);
    float ss = v.x * v.x + v.y * v.y;
    sdata[tid] = ss;
    __syncthreads();
    for (int off = 128; off > 0; off >>= 1) {
        if (tid < off) sdata[tid] += sdata[tid + off];
        __syncthreads();
    }
    const float scale = rsqrtf(sdata[0] * (1.0f / cfg::DM) + cfg::EPS);
    const float g0 = g[tid * 2 + 0];
    const float g1 = g[tid * 2 + 1];
    __bf16* o = out + (size_t)row * cfg::DM + tid * 2;
    o[0] = (__bf16)(v.x * scale * g0);
    o[1] = (__bf16)(v.y * scale * g1);
}

// ---------------------------------------------------------------------------
// Tiled WMMA GEMM with double-buffered LDS staging via async global->LDS.
//   C[M,N] = A[M,K](bf16 row-major) x Bt[N,K](bf16)^T
// block = 256 threads = 8 waves; block tile 128x64; K-step 64.
// ---------------------------------------------------------------------------
__global__ __launch_bounds__(256)
void gemm_bf16_kernel(const __bf16* __restrict__ A,
                      const __bf16* __restrict__ Bt,
                      int M, int N, int K,
                      const float* __restrict__ bias,
                      const float* __restrict__ residual,
                      int do_relu,
                      float* __restrict__ outF,
                      __bf16* __restrict__ outBF) {
    __shared__ __bf16 Alds[2][128 * LDA];
    __shared__ __bf16 Blds[2][64 * LDA];

    const int tid  = threadIdx.x;
    const int lane = tid & 31;
    const int wave = tid >> 5;
    const int half = lane >> 4;
    const int l16  = lane & 15;

    const int blkM = blockIdx.x * 128;
    const int blkN = blockIdx.y * 64;

    auto issue_async = [&](int bufSel, int k0) {
#pragma unroll
        for (int i = 0; i < 4; ++i) {
            const int c   = tid + i * 256;
            const int row = c >> 3;
            const int col = c & 7;
            const uint64_t ga = (uint64_t)(uintptr_t)
                (A + (size_t)(blkM + row) * K + k0 + col * 8);
            const uint32_t la = (uint32_t)(uintptr_t)
                (&Alds[bufSel][row * LDA + col * 8]);
            asm volatile("global_load_async_to_lds_b128 %0, %1, off"
                         :: "v"(la), "v"(ga) : "memory");
        }
#pragma unroll
        for (int i = 0; i < 2; ++i) {
            const int c   = tid + i * 256;
            const int row = c >> 3;
            const int col = c & 7;
            const uint64_t ga = (uint64_t)(uintptr_t)
                (Bt + (size_t)(blkN + row) * K + k0 + col * 8);
            const uint32_t la = (uint32_t)(uintptr_t)
                (&Blds[bufSel][row * LDA + col * 8]);
            asm volatile("global_load_async_to_lds_b128 %0, %1, off"
                         :: "v"(la), "v"(ga) : "memory");
        }
    };

    v8f z = {};
    v8f acc[4] = {z, z, z, z};

    issue_async(0, 0);
    asm volatile("s_wait_asynccnt 0x0" ::: "memory");
    __syncthreads();

    int buf = 0;
    for (int k0 = 0; k0 < K; k0 += 64) {
        if (k0 + 64 < K) issue_async(buf ^ 1, k0 + 64);

#pragma unroll
        for (int kk = 0; kk < 64; kk += 32) {
            FragBF a;
            const __bf16* Ab = &Alds[buf][(wave * 16 + l16) * LDA + kk];
            a.u[0] = *(const uint4*)(Ab + half * 8);
            a.u[1] = *(const uint4*)(Ab + 16 + half * 8);
#pragma unroll
            for (int t = 0; t < 4; ++t) {
                const __bf16* Bb =
                    &Blds[buf][(t * 16 + l16) * LDA + kk + half * 8];
                FragBF b;
                b.u[0] = *(const uint4*)(Bb);
                b.u[1] = *(const uint4*)(Bb + 16);
                acc[t] = __builtin_amdgcn_wmma_f32_16x16x32_bf16(
                    false, a.v, false, b.v, (short)0, acc[t], false, false);
            }
        }

        asm volatile("s_wait_asynccnt 0x0" ::: "memory");
        __syncthreads();
        buf ^= 1;
    }

    const int tileM = blkM + wave * 16;
#pragma unroll
    for (int t = 0; t < 4; ++t) {
#pragma unroll
        for (int r = 0; r < 8; ++r) {
            const int m = tileM + half * 8 + r;
            const int n = blkN + t * 16 + l16;
            float val = acc[t][r];
            if (bias)     val += bias[n];
            if (residual) val += residual[(size_t)m * N + n];
            if (do_relu)  val = fmaxf(val, 0.0f);
            if (outF)  outF [(size_t)m * N + n] = val;
            if (outBF) outBF[(size_t)m * N + n] = (__bf16)val;
        }
    }
}

// ---------------------------------------------------------------------------
// Random feature map: p = scale * (q . omega_h) ; out = [sin(p), cos(p)]/8
// qk: [B*S, H*64] f32 ; out: [B,H,S,64] bf16
// ---------------------------------------------------------------------------
__global__ __launch_bounds__(256)
void phi_kernel(const float* __restrict__ qk,
                const float* __restrict__ omega,
                __bf16* __restrict__ out,
                float qscale) {
    __shared__ float qs[8 * 64];
    const int p0 = blockIdx.x * 8;
    const int bh = p0 / cfg::S;             // constant within block (S % 8 == 0)
    const int s0 = p0 - bh * cfg::S;
    const int b  = bh / cfg::H;
    const int h  = bh - b * cfg::H;

    for (int idx = threadIdx.x; idx < 8 * 64; idx += 256) {
        const int lp = idx >> 6;
        const int d  = idx & 63;
        qs[idx] = qk[((size_t)(b * cfg::S + s0 + lp)) * cfg::DM + h * 64 + d];
    }
    __syncthreads();

    const int w = threadIdx.x >> 5;
    const int f = threadIdx.x & 31;
    float acc = 0.0f;
#pragma unroll 4
    for (int d = 0; d < 64; ++d)
        acc += qs[w * 64 + d] * omega[(size_t)(h * 64 + d) * 32 + f];
    const float p = qscale * acc;
    __bf16* o = out + ((size_t)bh * cfg::S + s0 + w) * cfg::F;
    o[f]      = (__bf16)(__sinf(p) * 0.125f);
    o[32 + f] = (__bf16)(__cosf(p) * 0.125f);
}

// ---------------------------------------------------------------------------
// Chunked causal linear attention, one workgroup per (b,h), chunk C = 64.
//
// Per chunk (all WMMA on 64x64x64, 8 waves x 2 tiles):
//   A      = Phi_q . Phi_k^T          (causal mask incl. diagonal -> Am bf16)
//   num    = Am . v  +  Phi_q . KV^T
//   den[j] = Phi_q[j] . ksum  +  rowsum(Am[j])
//   KV^T  += (Phi_k^T . v)^T          (accumulated in persistent wave regs)
// Phi_q/Phi_k chunks stream in via async global->LDS (contiguous 8KB tiles).
// ---------------------------------------------------------------------------
__global__ __launch_bounds__(256)
void rfa_scan_wmma_kernel(const __bf16* __restrict__ pq,
                          const __bf16* __restrict__ pk,
                          const __bf16* __restrict__ vbf,
                          __bf16* __restrict__ attnC) {
    __shared__ __bf16 Pq [64 * LDA];   // [s][f]
    __shared__ __bf16 Pk [64 * LDA];   // [s][f]
    __shared__ __bf16 PkT[64 * LDA];   // [f][s]
    __shared__ __bf16 Vt [64 * LDA];   // [e][s]  (v transposed)
    __shared__ __bf16 Am [64 * LDA];   // masked A, [j][i]
    __shared__ __bf16 KVbf[64 * LDA];  // KV^T bf16, [e][f]
    __shared__ float  ksum[64], D1[64], den[64];

    const int tid  = threadIdx.x;
    const int lane = tid & 31;
    const int wave = tid >> 5;
    const int half = lane >> 4;
    const int l16  = lane & 15;
    const int tr   = wave >> 1;          // tile row (0..3)
    const int tc0  = (wave & 1) * 2;     // first tile col (0 or 2)

    const int bh = blockIdx.x;
    const int b  = bh / cfg::H;
    const int h  = bh - b * cfg::H;

    const size_t pqBase = (size_t)bh * cfg::S * cfg::F;           // bf16 elems
    const size_t vBase  = (size_t)b * cfg::S * cfg::DM + h * 64;  // bf16 elems

    if (tid < 64) ksum[tid] = 0.0f;

    // persistent KV^T state tiles in registers: element (r) of kv[t]:
    //   f = tr*16 + half*8 + r ; e = (tc0+t)*16 + l16
    v8f z = {};
    v8f kv[2] = {z, z};
    __syncthreads();

    for (int c0 = 0; c0 < cfg::S; c0 += 64) {
        // ---- stage: async Phi chunks, v load+transpose, KVbf from regs ----
#pragma unroll
        for (int i = 0; i < 2; ++i) {
            const int c   = tid + i * 256;          // 512 chunks of 16B
            const int row = c >> 3;
            const int col = c & 7;
            {
                const uint64_t ga = (uint64_t)(uintptr_t)
                    (pq + pqBase + (size_t)c0 * 64 + c * 8);
                const uint32_t la = (uint32_t)(uintptr_t)
                    (&Pq[row * LDA + col * 8]);
                asm volatile("global_load_async_to_lds_b128 %0, %1, off"
                             :: "v"(la), "v"(ga) : "memory");
            }
            {
                const uint64_t ga = (uint64_t)(uintptr_t)
                    (pk + pqBase + (size_t)c0 * 64 + c * 8);
                const uint32_t la = (uint32_t)(uintptr_t)
                    (&PkT[0]);  // placeholder overwritten below? no -- use Pk
                (void)la;
                const uint32_t lp = (uint32_t)(uintptr_t)
                    (&Pk[row * LDA + col * 8]);
                asm volatile("global_load_async_to_lds_b128 %0, %1, off"
                             :: "v"(lp), "v"(ga) : "memory");
            }
        }
        // v: load 8 bf16 and scatter transposed into Vt
#pragma unroll
        for (int i = 0; i < 2; ++i) {
            const int c   = tid + i * 256;
            const int row = c >> 3;                 // s within chunk
            const int col = c & 7;                  // e group
            union { uint4 q; __bf16 e[8]; } raw;
            raw.q = *(const uint4*)(vbf + vBase + (size_t)(c0 + row) * cfg::DM + col * 8);
#pragma unroll
            for (int j = 0; j < 8; ++j)
                Vt[(col * 8 + j) * LDA + row] = raw.e[j];
        }
        // publish KV^T (bf16) from persistent registers
#pragma unroll
        for (int t = 0; t < 2; ++t)
#pragma unroll
            for (int r = 0; r < 8; ++r) {
                const int f = tr * 16 + half * 8 + r;
                const int e = (tc0 + t) * 16 + l16;
                KVbf[e * LDA + f] = (__bf16)kv[t][r];
            }
        asm volatile("s_wait_asynccnt 0x0" ::: "memory");
        __syncthreads();

        // ---- build Pk^T ; D1[j] = Phi_q[j] . ksum ----
        for (int i = tid; i < 4096; i += 256) {
            const int s = i >> 6, f = i & 63;
            PkT[f * LDA + s] = Pk[s * LDA + f];
        }
        if (tid < 64) {
            float d = 0.0f;
#pragma unroll 8
            for (int f = 0; f < 64; ++f)
                d += (float)Pq[tid * LDA + f] * ksum[f];
            D1[tid] = d;
        }
        __syncthreads();

        // ---- GEMM1: A = Pq . Pk^T ; causal mask ; Am (bf16) ----
        {
            v8f a2[2] = {z, z};
            wmma64(Pq, Pk, tr, tc0, half, l16, a2);
#pragma unroll
            for (int t = 0; t < 2; ++t)
#pragma unroll
                for (int r = 0; r < 8; ++r) {
                    const int m = tr * 16 + half * 8 + r;
                    const int n = (tc0 + t) * 16 + l16;
                    float val = a2[t][r];
                    if (n > m) val = 0.0f;
                    Am[m * LDA + n] = (__bf16)val;
                }
        }
        __syncthreads();

        // ---- GEMM2: num = Am . v + Pq . KV^T ; den ----
        v8f nacc[2] = {z, z};
        wmma64(Am, Vt, tr, tc0, half, l16, nacc);
        wmma64(Pq, KVbf, tr, tc0, half, l16, nacc);
        if (tid < 64) {
            float d = D1[tid];
#pragma unroll 8
            for (int i = 0; i < 64; ++i)
                d += (float)Am[tid * LDA + i];
            den[tid] = d;
        }
        __syncthreads();

        // ---- epilogue: attn ; GEMM4 state update ; ksum update ----
#pragma unroll
        for (int t = 0; t < 2; ++t)
#pragma unroll
            for (int r = 0; r < 8; ++r) {
                const int m = tr * 16 + half * 8 + r;     // j within chunk
                const int n = (tc0 + t) * 16 + l16;       // e
                const float val = nacc[t][r] / (den[m] + cfg::EPS);
                attnC[vBase + (size_t)(c0 + m) * cfg::DM + n] = (__bf16)val;
            }
        wmma64(PkT, Vt, tr, tc0, half, l16, kv);   // KV^T += Pk^T . v
        if (tid < 64) {
            float ka = 0.0f;
#pragma unroll 8
            for (int s = 0; s < 64; ++s)
                ka += (float)PkT[tid * LDA + s];
            ksum[tid] += ka;
        }
        __syncthreads();
    }
}

// ---------------------------------------------------------------------------
// Host-side orchestration
// ---------------------------------------------------------------------------
extern "C" void kernel_launch(void* const* d_in, const int* in_sizes, int n_in,
                              void* d_out, int out_size, void* d_ws, size_t ws_size,
                              hipStream_t stream) {
    (void)in_sizes; (void)n_in; (void)out_size; (void)ws_size;
    using namespace cfg;

    const float* x     = (const float*)d_in[0];
    const float* Wq    = (const float*)d_in[1];
    const float* Wk    = (const float*)d_in[2];
    const float* Wv    = (const float*)d_in[3];
    const float* Wo    = (const float*)d_in[4];
    const float* omega = (const float*)d_in[5];
    const float* g1    = (const float*)d_in[6];
    const float* g2    = (const float*)d_in[7];
    const float* W1    = (const float*)d_in[8];
    const float* b1    = (const float*)d_in[9];
    const float* W2    = (const float*)d_in[10];
    const float* b2    = (const float*)d_in[11];
    float* out = (float*)d_out;

    char*  ws  = (char*)d_ws;
    size_t off = 0;
    auto alloc = [&](size_t bytes) -> void* {
        void* p = ws + off;
        off = (off + bytes + 255) & ~(size_t)255;
        return p;
    };

    __bf16* xn_bf  = (__bf16*)alloc((size_t)M * DM * 2);
    __bf16* Wq_t   = (__bf16*)alloc((size_t)DM * DM * 2);
    __bf16* Wk_t   = (__bf16*)alloc((size_t)DM * DM * 2);
    __bf16* Wv_t   = (__bf16*)alloc((size_t)DM * DM * 2);
    __bf16* Wo_t   = (__bf16*)alloc((size_t)DM * DM * 2);
    __bf16* W1_t   = (__bf16*)alloc((size_t)DM * HID * 2);
    __bf16* W2_t   = (__bf16*)alloc((size_t)HID * DM * 2);
    float*  qf     = (float*)alloc((size_t)M * DM * 4);
    float*  kf     = (float*)alloc((size_t)M * DM * 4);
    __bf16* v_bf   = (__bf16*)alloc((size_t)M * DM * 2);
    __bf16* pq_bf  = (__bf16*)alloc((size_t)B * H * S * F * 2);
    __bf16* pk_bf  = (__bf16*)alloc((size_t)B * H * S * F * 2);
    __bf16* attnC  = (__bf16*)alloc((size_t)M * DM * 2);
    float*  yf     = (float*)alloc((size_t)M * DM * 4);
    __bf16* xn2_bf = (__bf16*)alloc((size_t)M * DM * 2);
    __bf16* h_bf   = (__bf16*)alloc((size_t)M * HID * 2);

    cast_transpose_kernel<<<(DM * DM) / 256, 256, 0, stream>>>(Wq, Wq_t, DM, DM);
    cast_transpose_kernel<<<(DM * DM) / 256, 256, 0, stream>>>(Wk, Wk_t, DM, DM);
    cast_transpose_kernel<<<(DM * DM) / 256, 256, 0, stream>>>(Wv, Wv_t, DM, DM);
    cast_transpose_kernel<<<(DM * DM) / 256, 256, 0, stream>>>(Wo, Wo_t, DM, DM);
    cast_transpose_kernel<<<(DM * HID) / 256, 256, 0, stream>>>(W1, W1_t, DM, HID);
    cast_transpose_kernel<<<(HID * DM) / 256, 256, 0, stream>>>(W2, W2_t, HID, DM);

    rmsnorm_kernel<<<M, 256, 0, stream>>>(x, g1, xn_bf);

    {
        dim3 grid(M / 128, DM / 64);
        gemm_bf16_kernel<<<grid, 256, 0, stream>>>(xn_bf, Wq_t, M, DM, DM,
                                                   nullptr, nullptr, 0, qf, nullptr);
        gemm_bf16_kernel<<<grid, 256, 0, stream>>>(xn_bf, Wk_t, M, DM, DM,
                                                   nullptr, nullptr, 0, kf, nullptr);
        gemm_bf16_kernel<<<grid, 256, 0, stream>>>(xn_bf, Wv_t, M, DM, DM,
                                                   nullptr, nullptr, 0, nullptr, v_bf);
    }

    {
        const float qscale = 0.35355339059327373f; // 64^-0.25
        const int nblocks = (B * H * S) / 8;
        phi_kernel<<<nblocks, 256, 0, stream>>>(qf, omega, pq_bf, qscale);
        phi_kernel<<<nblocks, 256, 0, stream>>>(kf, omega, pk_bf, qscale);
    }

    rfa_scan_wmma_kernel<<<B * H, 256, 0, stream>>>(pq_bf, pk_bf, v_bf, attnC);

    {
        dim3 grid(M / 128, DM / 64);
        gemm_bf16_kernel<<<grid, 256, 0, stream>>>(attnC, Wo_t, M, DM, DM,
                                                   nullptr, x, 0, yf, nullptr);
    }

    rmsnorm_kernel<<<M, 256, 0, stream>>>(yf, g2, xn2_bf);

    {
        dim3 grid(M / 128, HID / 64);
        gemm_bf16_kernel<<<grid, 256, 0, stream>>>(xn2_bf, W1_t, M, HID, DM,
                                                   b1, nullptr, 1, nullptr, h_bf);
    }

    {
        dim3 grid(M / 128, DM / 64);
        gemm_bf16_kernel<<<grid, 256, 0, stream>>>(h_bf, W2_t, M, DM, HID,
                                                   b2, yf, 0, out, nullptr);
    }
}
